// LearnedRouting_64982855188718
// MI455X (gfx1250) — matrix operand order repaired and checked by reading, not compiled, not run on hardware
//
#include <hip/hip_runtime.h>
#include <cstdint>

// Problem constants (match reference)
constexpr int NN = 8192;      // nodes
constexpr int BB = 2;         // batch
constexpr int TT = 1024;      // time steps
constexpr int WW = 64;        // IRF window
constexpr int HH = 256;       // hidden
constexpr float DTc = 1.0f / 24.0f;

typedef __attribute__((ext_vector_type(16))) _Float16 v16h;
typedef __attribute__((ext_vector_type(8)))  float    v8f;

// --- CDNA5 async global->LDS path (probe-gated so compile cannot fail) -----
#if defined(__has_builtin)
#  if __has_builtin(__builtin_amdgcn_global_load_async_to_lds_b128)
#    define USE_ASYNC_LDS 1
#  endif
#endif
#ifndef USE_ASYNC_LDS
#  define USE_ASYNC_LDS 0
#endif

// Builtin signature (from round-2 diagnostic): arg0 is
//   "__attribute__((vector_size(16))) int __device__ *"  i.e. v4i in AS1.
typedef int v4i __attribute__((vector_size(16)));
typedef __attribute__((address_space(1))) v4i gas_v4i;   // global (AS1)
typedef __attribute__((address_space(3))) v4i las_v4i;   // LDS (AS3)

__device__ __forceinline__ void wait_asynccnt0() {
#if defined(__has_builtin)
#  if __has_builtin(__builtin_amdgcn_s_wait_asynccnt)
    __builtin_amdgcn_s_wait_asynccnt(0);
    return;
#  endif
#endif
#if USE_ASYNC_LDS
    asm volatile("s_wait_asynccnt 0x0" ::: "memory");
#endif
}

// ---------------------------------------------------------------------------
// Stage 1a: layer1  h1 = relu(phys @ w1 + b1)  -> f16, row-major [N][H]
// ---------------------------------------------------------------------------
__global__ void mlp_l1_kernel(const float* __restrict__ phys,
                              const float* __restrict__ w1,
                              const float* __restrict__ b1,
                              _Float16* __restrict__ h1) {
    int idx = blockIdx.x * blockDim.x + threadIdx.x;   // N*H threads
    if (idx >= NN * HH) return;
    int n = idx / HH;
    int h = idx % HH;
    float v = b1[h];
    v = fmaf(phys[n * 3 + 0], w1[0 * HH + h], v);
    v = fmaf(phys[n * 3 + 1], w1[1 * HH + h], v);
    v = fmaf(phys[n * 3 + 2], w1[2 * HH + h], v);
    v = fmaxf(v, 0.0f);
    h1[idx] = (_Float16)v;
}

// ---------------------------------------------------------------------------
// Stage 1b: transpose+convert w2 (f32 [K][H]) -> w2t (f16 [H][K])
// ---------------------------------------------------------------------------
__global__ void w2_transpose_kernel(const float* __restrict__ w2,
                                    _Float16* __restrict__ w2t) {
    int idx = blockIdx.x * blockDim.x + threadIdx.x;   // H*H threads
    if (idx >= HH * HH) return;
    int k = idx / HH;
    int h = idx % HH;
    w2t[h * HH + k] = (_Float16)w2[k * HH + h];
}

// ---------------------------------------------------------------------------
// Stage 1c: layer2 via WMMA  h2 = relu(h1 @ w2 + b2), f32 [N][H]
// One wave per 16x16 tile of h2; K=256 -> 8x v_wmma_f32_16x16x32_f16.
// (codegen-verified round 1: 8 v_wmma + b128 loads + s_wait_loadcnt pipeline)
// ---------------------------------------------------------------------------
__global__ void mlp_l2_wmma_kernel(const _Float16* __restrict__ h1,
                                   const _Float16* __restrict__ w2t,
                                   const float* __restrict__ b2,
                                   float* __restrict__ h2) {
    const int lane = threadIdx.x & 31;
    const int wave = blockIdx.x * (blockDim.x >> 5) + (threadIdx.x >> 5);
    // (N/16) * (H/16) = 512 * 16 tiles
    const int tileH = wave & 15;
    const int tileN = wave >> 4;
    const int nbase = tileN * 16;
    const int hbase = tileH * 16;
    const int hf  = lane >> 4;     // 0 or 1 (lane half)
    const int l16 = lane & 15;

    const _Float16* arow = h1  + (size_t)(nbase + l16) * HH;  // A row M=l16
    const _Float16* brow = w2t + (size_t)(hbase + l16) * HH;  // B col=l16

    v8f c = {0.f, 0.f, 0.f, 0.f, 0.f, 0.f, 0.f, 0.f};

    for (int kb = 0; kb < HH; kb += 32) {
        union { v16h v; uint32_t u[8]; } A;
        #pragma unroll
        for (int r = 0; r < 8; ++r) {
            int K0 = hf * 16 + r * 2;   // even -> 4B aligned
            A.u[r] = *(const uint32_t*)(arow + kb + K0);
        }
        union { v16h v; uint4 q[2]; } Bm;
        const _Float16* bp = brow + kb + hf * 16;            // 32B aligned
        Bm.q[0] = *(const uint4*)(bp);
        Bm.q[1] = *(const uint4*)(bp + 8);

        c = __builtin_amdgcn_wmma_f32_16x16x32_f16(
                /*neg_a=*/false, A.v, /*neg_b=*/false, Bm.v,
                /*c_mod=*/(short)0, c, /*reuse_a=*/false, /*reuse_b=*/false);
    }

    #pragma unroll
    for (int r = 0; r < 8; ++r) {
        int M  = r + 8 * hf;
        int Nc = l16;
        float val = c[r] + b2[hbase + Nc];
        val = fmaxf(val, 0.0f);
        h2[(size_t)(nbase + M) * HH + hbase + Nc] = val;
    }
}

// ---------------------------------------------------------------------------
// Stage 1d: head (h2 @ w3 + b3) -> sigmoid params -> IRF[N][64]
// ---------------------------------------------------------------------------
__global__ void head_irf_kernel(const float* __restrict__ h2,
                                const float* __restrict__ w3,
                                const float* __restrict__ b3,
                                float* __restrict__ irf) {
    __shared__ float s_w3[2 * HH];
    const int tid = threadIdx.x;
    for (int i = tid; i < 2 * HH; i += blockDim.x) s_w3[i] = w3[i];
    __syncthreads();

    int n = blockIdx.x * blockDim.x + tid;
    if (n >= NN) return;

    float r0 = b3[0], r1 = b3[1];
    const float* hr = h2 + (size_t)n * HH;
    #pragma unroll 4
    for (int h = 0; h < HH; ++h) {
        float v = hr[h];
        r0 = fmaf(v, s_w3[2 * h + 0], r0);
        r1 = fmaf(v, s_w3[2 * h + 1], r1);
    }
    // p = sigmoid(raw - [0,3]) * [0.25,1.2] + [0.005,0]
    float kpar = 0.25f / (1.0f + __expf(-r0)) + 0.005f;
    float xw   = 1.2f  / (1.0f + __expf(-(r1 - 3.0f)));

    float delay = kpar * xw;
    float tau   = fmaxf(kpar * (1.0f - xw), 0.5f * DTc);
    float itau  = 1.0f / tau;

    float sum = 0.0f;
    #pragma unroll
    for (int w = 0; w < WW; ++w) {
        float t  = (float)w * DTc;
        float hv = (t >= delay) ? __expf(-(t - delay) * itau) * itau : 0.0f;
        sum += hv;
    }
    float inv = 1.0f / (sum + 1e-8f);
    #pragma unroll
    for (int w = 0; w < WW; ++w) {
        float t  = (float)w * DTc;
        float hv = (t >= delay) ? __expf(-(t - delay) * itau) * itau : 0.0f;
        irf[n * WW + w] = hv * inv;
    }
}

// ---------------------------------------------------------------------------
// Stage 2: routing, one level per launch.
// Block = (parent-group, batch): convolves <=4 children (contiguous ids
// 4p+1..4p+4), writes out[child], accumulates into acc[parent] (race-free:
// each parent owned by exactly one block).
//
// LDS tile layout (320 floats): index i <-> time t = t0 - 64 + i.
//   i in [0,63]   : halo (regular loads, zero-filled for t<0)
//   i in [64,319] : bulk in[t0 .. t0+255]; starts at LDS byte 256 (16B aligned)
//                   -> 64 lanes x GLOBAL_LOAD_ASYNC_TO_LDS_B128 on CDNA5
// Output y[t0+tid] = sum_w irf[w] * s_in[64 + tid - w].
// ---------------------------------------------------------------------------
__global__ void route_level_kernel(float* __restrict__ acc,
                                   float* __restrict__ out,
                                   const float* __restrict__ irf,
                                   int cstart, int cend, int pstart) {
    __shared__ float s_irf[WW];
    __shared__ float s_in[256 + WW];       // 320 floats

    const int tid = threadIdx.x;           // 256 threads
    const int b   = blockIdx.y;
    const int c0  = cstart + 4 * (int)blockIdx.x;
    const int nch = min(4, cend - c0);
    const int parent = (pstart < 0) ? -1 : (pstart + (int)blockIdx.x);

    float psum[4] = {0.f, 0.f, 0.f, 0.f};

    for (int ci = 0; ci < nch; ++ci) {
        const int c = c0 + ci;
        const float* in = acc + ((size_t)b * NN + c) * TT;
        float*       o  = out + ((size_t)b * NN + c) * TT;

        __syncthreads();                   // protect s_irf vs prev child reads
        if (tid < WW) s_irf[tid] = irf[c * WW + tid];

        for (int kk = 0; kk < TT / 256; ++kk) {
            const int t0 = kk * 256;
            __syncthreads();               // protect s_in reuse
#if USE_ASYNC_LDS
            if (tid < 64) {
                // halo: t = t0-64+tid
                int t = t0 - 64 + tid;
                s_in[tid] = (t >= 0) ? in[t] : 0.0f;
                // bulk: 4 dwords per lane, LDS offset 256+16*tid (16B aligned)
                const float* gsrc = in + t0 + 4 * tid;
                __builtin_amdgcn_global_load_async_to_lds_b128(
                    (gas_v4i*)gsrc,
                    (las_v4i*)&s_in[64 + 4 * tid],
                    /*offset=*/0, /*cpol=*/0);
            }
            wait_asynccnt0();
            __syncthreads();
#else
            for (int i = tid; i < 256 + WW; i += 256) {
                int t = t0 - WW + i;
                s_in[i] = (t >= 0) ? in[t] : 0.0f;
            }
            __syncthreads();
#endif
            float y = 0.0f;
            #pragma unroll
            for (int w = 0; w < WW; ++w)
                y = fmaf(s_irf[w], s_in[64 + tid - w], y);

            o[t0 + tid] = y;
            psum[kk] += y;
        }
    }

    if (parent >= 0) {
        float* pa = acc + ((size_t)b * NN + parent) * TT;
        #pragma unroll
        for (int kk = 0; kk < TT / 256; ++kk)
            pa[kk * 256 + tid] += psum[kk];
    }
}

// ---------------------------------------------------------------------------
// Launch
// ---------------------------------------------------------------------------
extern "C" void kernel_launch(void* const* d_in, const int* in_sizes, int n_in,
                              void* d_out, int out_size, void* d_ws, size_t ws_size,
                              hipStream_t stream) {
    const float* x    = (const float*)d_in[0];   // (B,N,T)
    const float* phys = (const float*)d_in[1];   // (N,3)
    const float* w1   = (const float*)d_in[2];   // (3,H)
    const float* b1   = (const float*)d_in[3];   // (H,)
    const float* w2   = (const float*)d_in[4];   // (H,H)
    const float* b2   = (const float*)d_in[5];   // (H,)
    const float* w3   = (const float*)d_in[6];   // (H,2)
    const float* b3   = (const float*)d_in[7];   // (2,)
    float* out = (float*)d_out;                  // (B,N,T)

    // Workspace layout
    char* ws = (char*)d_ws;
    const size_t ACC_BYTES = (size_t)BB * NN * TT * sizeof(float);      // 67,108,864
    const size_t H1_BYTES  = (size_t)NN * HH * sizeof(_Float16);        //  4,194,304
    const size_t W2T_BYTES = (size_t)HH * HH * sizeof(_Float16);        //    131,072
    const size_t H2_BYTES  = (size_t)NN * HH * sizeof(float);           //  8,388,608
    float*    acc  = (float*)   (ws);
    _Float16* h1   = (_Float16*)(ws + ACC_BYTES);
    _Float16* w2t  = (_Float16*)(ws + ACC_BYTES + H1_BYTES);
    float*    h2   = (float*)   (ws + ACC_BYTES + H1_BYTES + W2T_BYTES);
    float*    irf  = (float*)   (ws + ACC_BYTES + H1_BYTES + W2T_BYTES + H2_BYTES);

    // acc <- x
    (void)hipMemcpyAsync(acc, x, ACC_BYTES, hipMemcpyDeviceToDevice, stream);

    // MLP + IRF
    mlp_l1_kernel<<<(NN * HH + 255) / 256, 256, 0, stream>>>(phys, w1, b1, h1);
    w2_transpose_kernel<<<(HH * HH + 255) / 256, 256, 0, stream>>>(w2, w2t);
    {
        const int waves  = (NN / 16) * (HH / 16);     // 8192 tiles, 1 wave each
        const int blocks = waves / 8;                 // 8 waves (256 thr) / block
        mlp_l2_wmma_kernel<<<blocks, 256, 0, stream>>>(h1, w2t, b2, h2);
    }
    head_irf_kernel<<<(NN + 255) / 256, 256, 0, stream>>>(h2, w3, b3, irf);

    // Routing: depth-d node ranges are contiguous; s[d] = (4^d - 1)/3
    static const int s[9] = {0, 1, 5, 21, 85, 341, 1365, 5461, 21845};
    for (int d = 7; d >= 0; --d) {
        int cstart = s[d];
        int cend   = s[d + 1] < NN ? s[d + 1] : NN;
        int pstart = (d == 0) ? -1 : s[d - 1];
        int groups = (cend - cstart + 3) / 4;
        dim3 grid(groups, BB);
        route_level_kernel<<<grid, 256, 0, stream>>>(acc, out, irf, cstart, cend, pstart);
    }
}
